// GruHANModel_79465484911027
// MI455X (gfx1250) — matrix operand order repaired and checked by reading, not compiled, not run on hardware
//
#include <hip/hip_runtime.h>
#include <math.h>

// ---------------- model constants ----------------
#define HIDD   128
#define NHEAD  4
#define DHEAD  32
#define OUTF   24
#define TT     16
#define NNODE  6000
#define NEDGE  120000
#define FWAT   16
#define FCIT   16
#define FSTA   8
#define G3     384          // 3*HIDD

typedef __attribute__((ext_vector_type(16))) _Float16 v16h;
typedef __attribute__((ext_vector_type(8)))  _Float16 v8h;
typedef __attribute__((ext_vector_type(2)))  __fp16   v2fp16;  // cvt_pkrtz result type
typedef __attribute__((ext_vector_type(8)))  float    v8f;

#define ACT_NONE 0
#define ACT_RELU 1
#define ACT_SIGM 2

// =====================================================================
// Generic WMMA GEMM:  C[M,N] = act( (accum? C : 0) + A[M,K] * op(B) + bias )
// op(B): transB=1 -> B stored (N,K) rows (x @ W^T), transB=0 -> B (K,N).
// Block = 4 waves (128 thr). Wave tile = 16(M) x 32(N): one A fragment
// feeds two WMMAs. Block tile = 64 x 32.
// f32 -> f16 staged through LDS; fragments gathered with 16B ds loads.
// =====================================================================
__global__ __launch_bounds__(128)
void gemm_wmma_f16(const float* __restrict__ A, int lda,
                   const float* __restrict__ B, int ldb, int transB,
                   const float* __restrict__ bias,
                   float* __restrict__ C, int ldc,
                   int M, int N, int K, int accum, int act)
{
    __shared__ __align__(16) _Float16 As[4][16][32];   // [wave][m][k]
    __shared__ __align__(16) _Float16 Bs[32][32];      // [n][k]  (n-major!)

    const int tid  = threadIdx.x;
    const int w    = tid >> 5;          // wave in block
    const int lane = tid & 31;
    const int g    = lane >> 4;         // lane group (0/1)
    const int ml   = lane & 15;
    const int n0   = blockIdx.x * 32;
    const int m0   = blockIdx.y * 64 + w * 16;

    v8f acc0 = {}, acc1 = {};
    for (int k0 = 0; k0 < K; k0 += 32) {
        // ---- stage B tile 32n x 32k (whole block) ----
        if (transB && (n0 + 32 <= N) && (k0 + 32 <= K)) {
            #pragma unroll
            for (int it = 0; it < 4; ++it) {
                int s  = tid + 128 * it;           // dword-pair slot
                int n  = s >> 4, cp = s & 15;
                const float* bp = B + (size_t)(n0 + n) * ldb + k0 + 2 * cp;
                float2 f = *reinterpret_cast<const float2*>(bp);
                v2fp16 p = __builtin_amdgcn_cvt_pkrtz(f.x, f.y);
                *reinterpret_cast<v2fp16*>(&Bs[n][2 * cp]) = p;
            }
        } else {
            for (int s = tid; s < 1024; s += 128) {
                int n = s >> 5, kk = s & 31;
                float v = 0.f;
                if (n0 + n < N && k0 + kk < K)
                    v = transB ? B[(size_t)(n0 + n) * ldb + (k0 + kk)]
                               : B[(size_t)(k0 + kk) * ldb + (n0 + n)];
                Bs[n][kk] = (_Float16)v;
            }
        }
        // ---- stage A tile 16m x 32k (per wave) ----
        if ((m0 + 16 <= M) && (k0 + 32 <= K)) {
            #pragma unroll
            for (int it = 0; it < 8; ++it) {
                int s  = lane + 32 * it;
                int r  = s >> 4, cp = s & 15;
                const float* ap = A + (size_t)(m0 + r) * lda + k0 + 2 * cp;
                float2 f = *reinterpret_cast<const float2*>(ap);
                v2fp16 p = __builtin_amdgcn_cvt_pkrtz(f.x, f.y);
                *reinterpret_cast<v2fp16*>(&As[w][r][2 * cp]) = p;
                if (it == 0 && k0 + 32 < K) __builtin_prefetch(ap + 32, 0, 1);
            }
        } else {
            for (int s = lane; s < 512; s += 32) {
                int r = s >> 5, kk = s & 31;
                float v = 0.f;
                if (m0 + r < M && k0 + kk < K)
                    v = A[(size_t)(m0 + r) * lda + (k0 + kk)];
                As[w][r][kk] = (_Float16)v;
            }
        }
        __syncthreads();

        // ---- gather fragments (contiguous 16B LDS loads) ----
        // A frag, lane L: m=L%16, k = 16g..16g+15
        v8h alo = *reinterpret_cast<v8h*>(&As[w][ml][16 * g]);
        v8h ahi = *reinterpret_cast<v8h*>(&As[w][ml][16 * g + 8]);
        v16h av = __builtin_shufflevector(alo, ahi, 0,1,2,3,4,5,6,7,8,9,10,11,12,13,14,15);
        // B frag j, lane L: n=16j+L%16, k = 16g..16g+15
        v8h b0lo = *reinterpret_cast<v8h*>(&Bs[ml][16 * g]);
        v8h b0hi = *reinterpret_cast<v8h*>(&Bs[ml][16 * g + 8]);
        v16h bv0 = __builtin_shufflevector(b0lo, b0hi, 0,1,2,3,4,5,6,7,8,9,10,11,12,13,14,15);
        v8h b1lo = *reinterpret_cast<v8h*>(&Bs[16 + ml][16 * g]);
        v8h b1hi = *reinterpret_cast<v8h*>(&Bs[16 + ml][16 * g + 8]);
        v16h bv1 = __builtin_shufflevector(b1lo, b1hi, 0,1,2,3,4,5,6,7,8,9,10,11,12,13,14,15);

        acc0 = __builtin_amdgcn_wmma_f32_16x16x32_f16(false, av, false, bv0,
                                                      (short)0, acc0, false, false);
        acc1 = __builtin_amdgcn_wmma_f32_16x16x32_f16(false, av, false, bv1,
                                                      (short)0, acc1, false, false);
        __syncthreads();
    }

    // C/D layout: lane L: N = L%16, VGPR i -> M = i + 8*(L/16)
    #pragma unroll
    for (int i = 0; i < 8; ++i) {
        int m = m0 + i + 8 * g;
        if (m < M) {
            int n = n0 + ml;
            if (n < N) {
                float v = acc0[i];
                if (bias)  v += bias[n];
                if (accum) v += C[(size_t)m * ldc + n];
                if (act == ACT_RELU)      v = fmaxf(v, 0.f);
                else if (act == ACT_SIGM) v = 1.f / (1.f + expf(-v));
                C[(size_t)m * ldc + n] = v;
            }
            n = n0 + 16 + ml;
            if (n < N) {
                float v = acc1[i];
                if (bias)  v += bias[n];
                if (accum) v += C[(size_t)m * ldc + n];
                if (act == ACT_RELU)      v = fmaxf(v, 0.f);
                else if (act == ACT_SIGM) v = 1.f / (1.f + expf(-v));
                C[(size_t)m * ldc + n] = v;
            }
        }
    }
}

// =====================================================================
// Elementwise / utility kernels
// =====================================================================
__global__ void fill_kernel(float* p, float v, int n) {
    int i = blockIdx.x * 256 + threadIdx.x;
    if (i < n) p[i] = v;
}

// GRU gate combine: xw,hh are (N,384) with [r|z|n] column blocks (biases included)
__global__ void gru_combine_kernel(const float* __restrict__ xw,
                                   const float* __restrict__ hh,
                                   float* __restrict__ h,
                                   float* __restrict__ seq_t, int nh)
{
    int i = blockIdx.x * 256 + threadIdx.x;
    if (i >= nh) return;
    int n = i >> 7, c = i & 127;
    const float* xr = xw + (size_t)n * G3;
    const float* hr = hh + (size_t)n * G3;
    float r  = 1.f / (1.f + expf(-(xr[c]       + hr[c])));
    float z  = 1.f / (1.f + expf(-(xr[128 + c] + hr[128 + c])));
    float nn = tanhf(xr[256 + c] + r * hr[256 + c]);
    float h2 = (1.f - z) * nn + z * h[i];
    h[i] = h2;
    seq_t[i] = h2;
}

__global__ void gelu_kernel(const float* __restrict__ x, float* __restrict__ y, int n) {
    int i = blockIdx.x * 256 + threadIdx.x;
    if (i >= n) return;
    float v = x[i];
    y[i] = 0.5f * v * (1.f + erff(v * 0.70710678118f));
}

// HGT output skip + time-embedding add (x updated in place)
__global__ void hgt_skip_kernel(const float* __restrict__ o,
                                const float* __restrict__ skip,
                                const float* __restrict__ temb_row,
                                float* __restrict__ x, int n)
{
    int i = blockIdx.x * 256 + threadIdx.x;
    if (i >= n) return;
    float s = 1.f / (1.f + expf(-skip[0]));
    x[i] = s * o[i] + (1.f - s) * x[i] + temb_row[i & 127];
}

__global__ void gate_combine_kernel(const float* __restrict__ g,
                                    const float* __restrict__ hc,
                                    const float* __restrict__ hw,
                                    float* __restrict__ out, int n)
{
    int i = blockIdx.x * 256 + threadIdx.x;
    if (i >= n) return;
    out[i] = g[i] * hc[i] + (1.f - g[i]) * hw[i];
}

// LayerNorm over 128 features, one wave32 per row; optional residual add.
__global__ __launch_bounds__(32)
void ln_kernel(const float* __restrict__ x, const float* __restrict__ add,
               const float* __restrict__ gg, const float* __restrict__ bb,
               float* __restrict__ out, int rows)
{
    int r = blockIdx.x;
    if (r >= rows) return;
    int lane = threadIdx.x;
    float v[4]; float s = 0.f;
    #pragma unroll
    for (int i = 0; i < 4; ++i) {
        int c = lane + 32 * i;
        float t = x[(size_t)r * 128 + c];
        if (add) t += add[(size_t)r * 128 + c];
        v[i] = t; s += t;
    }
    for (int off = 16; off; off >>= 1) s += __shfl_xor(s, off);
    float mean = s * (1.f / 128.f);
    float vs = 0.f;
    #pragma unroll
    for (int i = 0; i < 4; ++i) { float d = v[i] - mean; vs += d * d; }
    for (int off = 16; off; off >>= 1) vs += __shfl_xor(vs, off);
    float inv = rsqrtf(vs * (1.f / 128.f) + 1e-5f);
    #pragma unroll
    for (int i = 0; i < 4; ++i) {
        int c = lane + 32 * i;
        out[(size_t)r * 128 + c] = (v[i] - mean) * inv * gg[c] + bb[c];
    }
}

// =====================================================================
// HGT edge kernels (segment softmax + aggregation via global float atomics)
// =====================================================================
__device__ __forceinline__ void atomicMaxF(float* addr, float v) {
    if (v >= 0.f) atomicMax((int*)addr, __float_as_int(v));
    else          atomicMin((unsigned int*)addr, (unsigned int)__float_as_int(v));
}

__global__ void edge_score_kernel(const int* __restrict__ src, const int* __restrict__ dst,
                                  const float* __restrict__ q, const float* __restrict__ kt,
                                  const float* __restrict__ mu,
                                  float* __restrict__ score, float* __restrict__ smax, int E)
{
    int idx = blockIdx.x * 256 + threadIdx.x;
    if (idx >= E * NHEAD) return;
    int e = idx >> 2, h = idx & 3;
    int sN = src[e], dN = dst[e];
    const float* qp = q  + (size_t)dN * HIDD + h * DHEAD;
    const float* kp = kt + (size_t)sN * HIDD + h * DHEAD;
    float a = 0.f;
    #pragma unroll 8
    for (int d = 0; d < DHEAD; ++d) a += qp[d] * kp[d];
    a *= mu[h] * 0.17677669529f;   // 1/sqrt(32)
    score[idx] = a;
    atomicMaxF(&smax[dN * NHEAD + h], a);
}

__global__ void edge_exp_kernel(const int* __restrict__ dst,
                                float* __restrict__ score,
                                const float* __restrict__ smax,
                                float* __restrict__ den, int E)
{
    int idx = blockIdx.x * 256 + threadIdx.x;
    if (idx >= E * NHEAD) return;
    int e = idx >> 2, h = idx & 3;
    int dN = dst[e];
    float a = expf(score[idx] - smax[dN * NHEAD + h]);
    score[idx] = a;
    atomicAdd(&den[dN * NHEAD + h], a);
}

__global__ void edge_agg_kernel(const int* __restrict__ src, const int* __restrict__ dst,
                                const float* __restrict__ score, const float* __restrict__ den,
                                const float* __restrict__ mt, float* __restrict__ agg, int E)
{
    int idx = blockIdx.x * 256 + threadIdx.x;
    if (idx >= E * HIDD) return;
    int e = idx >> 7, c = idx & 127, h = c >> 5;
    int sN = src[e], dN = dst[e];
    float w = score[e * NHEAD + h] / (den[dN * NHEAD + h] + 1e-16f);
    atomicAdd(&agg[(size_t)dN * HIDD + c], mt[(size_t)sN * HIDD + c] * w);
}

// =====================================================================
// Temporal self-attention: block(256) per (node, head); T=16, d=32 in LDS
// qkv layout: row r=(t*N+n), 384 cols (q|k|v)
// =====================================================================
__global__ __launch_bounds__(256)
void tattn_kernel(const float* __restrict__ qkv, float* __restrict__ attn, int Nn)
{
    int n = blockIdx.x, h = blockIdx.y, tid = threadIdx.x;
    __shared__ float qs[TT][DHEAD], ks[TT][DHEAD], vs[TT][DHEAD], sc[TT][TT + 1];
    for (int j = tid; j < TT * DHEAD; j += 256) {
        int t = j >> 5, d = j & 31;
        size_t base = ((size_t)t * Nn + n) * G3 + h * DHEAD + d;
        qs[t][d] = qkv[base];
        ks[t][d] = qkv[base + 128];
        vs[t][d] = qkv[base + 256];
    }
    __syncthreads();
    {
        int tq = tid >> 4, tk = tid & 15;     // 256 threads == 16x16 scores
        float a = 0.f;
        #pragma unroll 8
        for (int d = 0; d < DHEAD; ++d) a += qs[tq][d] * ks[tk][d];
        sc[tq][tk] = a * 0.17677669529f;
    }
    __syncthreads();
    if (tid < TT) {
        float m = -1e30f;
        for (int k = 0; k < TT; ++k) m = fmaxf(m, sc[tid][k]);
        float s = 0.f;
        for (int k = 0; k < TT; ++k) { float e = expf(sc[tid][k] - m); sc[tid][k] = e; s += e; }
        float inv = 1.f / s;
        for (int k = 0; k < TT; ++k) sc[tid][k] *= inv;
    }
    __syncthreads();
    for (int j = tid; j < TT * DHEAD; j += 256) {
        int tq = j >> 5, d = j & 31;
        float a = 0.f;
        #pragma unroll
        for (int k = 0; k < TT; ++k) a += sc[tq][k] * vs[k][d];
        attn[((size_t)tq * Nn + n) * HIDD + h * DHEAD + d] = a;
    }
}

// =====================================================================
// Cross attention (1 x 6000 queries over 6000 keys), online softmax.
// One wave per (query, head); lane = head dim.
// =====================================================================
__global__ __launch_bounds__(32)
void cross_attn_kernel(const float* __restrict__ qx, const float* __restrict__ kx,
                       const float* __restrict__ vx, float* __restrict__ o, int Nk)
{
    int qi = blockIdx.x, h = blockIdx.y, lane = threadIdx.x;
    float qd = qx[(size_t)qi * HIDD + h * DHEAD + lane];
    float m = -1e30f, l = 0.f, acc = 0.f;
    for (int j = 0; j < Nk; ++j) {
        float p = qd * kx[(size_t)j * HIDD + h * DHEAD + lane];
        for (int off = 16; off; off >>= 1) p += __shfl_xor(p, off);
        float s  = p * 0.17677669529f;
        float mn = fmaxf(m, s);
        float cs = expf(m - mn);
        float w  = expf(s - mn);
        l   = l * cs + w;
        acc = acc * cs + w * vx[(size_t)j * HIDD + h * DHEAD + lane];
        m = mn;
    }
    o[(size_t)qi * HIDD + h * DHEAD + lane] = acc / (l + 1e-20f);
}

// =====================================================================
// Host orchestration
// =====================================================================
static void gemm(hipStream_t s, const float* A, int lda, const float* B, int ldb, int transB,
                 const float* bias, float* C, int ldc, int M, int Nd, int K, int accum, int act)
{
    dim3 grid((Nd + 31) / 32, (M + 63) / 64);
    gemm_wmma_f16<<<grid, 128, 0, s>>>(A, lda, B, ldb, transB, bias, C, ldc, M, Nd, K, accum, act);
}
static void fillf(hipStream_t s, float* p, float v, size_t n) {
    fill_kernel<<<(unsigned)((n + 255) / 256), 256, 0, s>>>(p, v, (int)n);
}

static void attend(hipStream_t s, const int* src, const int* dst,
                   const float* q, const float* kt, const float* mt, const float* mu,
                   float* score, float* smax, float* den, float* agg)
{
    fillf(s, smax, -3.0e38f, (size_t)NNODE * NHEAD);
    fillf(s, den, 0.f, (size_t)NNODE * NHEAD);
    fillf(s, agg, 0.f, (size_t)NNODE * HIDD);
    int eh = NEDGE * NHEAD;
    edge_score_kernel<<<(eh + 255) / 256, 256, 0, s>>>(src, dst, q, kt, mu, score, smax, NEDGE);
    edge_exp_kernel<<<(eh + 255) / 256, 256, 0, s>>>(dst, score, smax, den, NEDGE);
    int ec = NEDGE * HIDD;
    edge_agg_kernel<<<(ec + 255) / 256, 256, 0, s>>>(src, dst, score, den, mt, agg, NEDGE);
}

extern "C" void kernel_launch(void* const* d_in, const int* in_sizes, int n_in,
                              void* d_out, int out_size, void* d_ws, size_t ws_size,
                              hipStream_t stream)
{
    (void)in_sizes; (void)n_in; (void)out_size; (void)ws_size;
    // ---- parameter leaves, setup_inputs() insertion order ----
    const float* gw_Wx = (const float*)d_in[0];
    const float* gw_Wh = (const float*)d_in[1];
    const float* gw_bx = (const float*)d_in[2];
    const float* gw_bh = (const float*)d_in[3];
    const float* gc_Wx = (const float*)d_in[4];
    const float* gc_Wh = (const float*)d_in[5];
    const float* gc_bx = (const float*)d_in[6];
    const float* gc_bh = (const float*)d_in[7];
    const float* fz_W  = (const float*)d_in[8];
    const float* fz_b  = (const float*)d_in[9];
    const float* lw_kW = (const float*)d_in[10]; const float* lw_kb = (const float*)d_in[11];
    const float* lw_qW = (const float*)d_in[12]; const float* lw_qb = (const float*)d_in[13];
    const float* lw_vW = (const float*)d_in[14]; const float* lw_vb = (const float*)d_in[15];
    const float* lw_aW = (const float*)d_in[16]; const float* lw_ab = (const float*)d_in[17];
    const float* lw_sk = (const float*)d_in[18];
    const float* lc_kW = (const float*)d_in[19]; const float* lc_kb = (const float*)d_in[20];
    const float* lc_qW = (const float*)d_in[21]; const float* lc_qb = (const float*)d_in[22];
    const float* lc_vW = (const float*)d_in[23]; const float* lc_vb = (const float*)d_in[24];
    const float* lc_aW = (const float*)d_in[25]; const float* lc_ab = (const float*)d_in[26];
    const float* lc_sk = (const float*)d_in[27];
    const float* rwc_att = (const float*)d_in[28];
    const float* rwc_msg = (const float*)d_in[29];
    const float* rwc_mu  = (const float*)d_in[30];
    const float* rcw_att = (const float*)d_in[31];
    const float* rcw_msg = (const float*)d_in[32];
    const float* rcw_mu  = (const float*)d_in[33];
    const float* temb  = (const float*)d_in[34];
    const float* ta_Wi = (const float*)d_in[35]; const float* ta_bi = (const float*)d_in[36];
    const float* ta_Wo = (const float*)d_in[37]; const float* ta_bo = (const float*)d_in[38];
    const float* xa_Wi = (const float*)d_in[39]; const float* xa_bi = (const float*)d_in[40];
    const float* xa_Wo = (const float*)d_in[41]; const float* xa_bo = (const float*)d_in[42];
    const float* ffn_W1 = (const float*)d_in[43]; const float* ffn_b1 = (const float*)d_in[44];
    const float* ffn_W2 = (const float*)d_in[45]; const float* ffn_b2 = (const float*)d_in[46];
    const float* gt_W  = (const float*)d_in[47]; const float* gt_b = (const float*)d_in[48];
    const float* n1_g = (const float*)d_in[49]; const float* n1_b = (const float*)d_in[50];
    const float* n2_g = (const float*)d_in[51]; const float* n2_b = (const float*)d_in[52];
    const float* n3_g = (const float*)d_in[53]; const float* n3_b = (const float*)d_in[54];
    const float* pr_W1 = (const float*)d_in[55]; const float* pr_b1 = (const float*)d_in[56];
    const float* pr_W2 = (const float*)d_in[57]; const float* pr_b2 = (const float*)d_in[58];
    const float* water_x  = (const float*)d_in[59];
    const float* city_x   = (const float*)d_in[60];
    const float* city_st  = (const float*)d_in[61];
    const int*   e_wc     = (const int*)d_in[62];
    const int*   e_cw     = (const int*)d_in[63];

    const int   Nn  = NNODE;
    const size_t NH  = (size_t)Nn * HIDD;
    const size_t TNH = (size_t)TT * NH;

    // ---- workspace partition (floats) ----
    float* W = (float*)d_ws;
    size_t off = 0;
    auto take = [&](size_t n) { float* p = W + off; off += n; return p; };
    float* seqW   = take(TNH);                    // water hidden seq (T,N,H), later hw_t / hw_tmp
    float* fusedC = take(TNH);                    // city fused seq, later hc_t / hc_tmp
    float* seqC   = take(TNH);                    // raw GRU city seq; REUSED as MHA attn buffer
    float* qkv    = take((size_t)TT * Nn * G3);   // MHA qkv; first TNH floats REUSED as proj tmp
    float* xwT    = take((size_t)Nn * G3);
    float* hhT    = take((size_t)Nn * G3);
    float* hst    = take(NH);
    float* Pkw = take(NH); float* Pvw = take(NH); float* Pqw = take(NH);
    float* Pkc = take(NH); float* Pvc = take(NH); float* Pqc = take(NH);
    float* KTw = take(NH); float* MTw = take(NH);
    float* KTc = take(NH); float* MTc = take(NH);
    float* scoreB = take((size_t)NEDGE * NHEAD);
    float* smaxB  = take((size_t)Nn * NHEAD);
    float* denB   = take((size_t)Nn * NHEAD);
    float* aggB = take(NH); float* gelB = take(NH); float* oT = take(NH);
    float* qx = take(NH); float* kx = take(NH); float* vx = take(NH);
    float* crs = take(NH); float* hcx = take(NH);
    float* gB = take(NH);  float* hB = take(NH);
    float* f1 = take((size_t)Nn * 2 * HIDD); float* f2 = take(NH);
    float* p1 = take(NH);

    // ================= GRU water =================
    fillf(stream, hst, 0.f, NH);
    for (int t = 0; t < TT; ++t) {
        gemm(stream, water_x + t * FWAT, TT * FWAT, gw_Wx, FWAT, 1, gw_bx, xwT, G3, Nn, G3, FWAT, 0, ACT_NONE);
        gemm(stream, hst, HIDD, gw_Wh, HIDD, 1, gw_bh, hhT, G3, Nn, G3, HIDD, 0, ACT_NONE);
        gru_combine_kernel<<<(unsigned)((NH + 255) / 256), 256, 0, stream>>>(xwT, hhT, hst, seqW + t * NH, (int)NH);
    }
    // ================= GRU city =================
    fillf(stream, hst, 0.f, NH);
    for (int t = 0; t < TT; ++t) {
        gemm(stream, city_x + t * FCIT, TT * FCIT, gc_Wx, FCIT, 1, gc_bx, xwT, G3, Nn, G3, FCIT, 0, ACT_NONE);
        gemm(stream, hst, HIDD, gc_Wh, HIDD, 1, gc_bh, hhT, G3, Nn, G3, HIDD, 0, ACT_NONE);
        gru_combine_kernel<<<(unsigned)((NH + 255) / 256), 256, 0, stream>>>(xwT, hhT, hst, seqC + t * NH, (int)NH);
    }
    // ================= fusion: relu([h_city|static] @ W^T + b) =================
    for (int t = 0; t < TT; ++t) {
        gemm(stream, seqC + t * NH, HIDD, fz_W, HIDD + FSTA, 1, fz_b, fusedC + t * NH, HIDD, Nn, HIDD, HIDD, 0, ACT_NONE);
        gemm(stream, city_st, FSTA, fz_W + HIDD, HIDD + FSTA, 1, nullptr, fusedC + t * NH, HIDD, Nn, HIDD, FSTA, 1, ACT_RELU);
    }
    // ================= HGT over time (in-place into seqW / fusedC, +temb) ======
    const int* wc_src = e_wc;          const int* wc_dst = e_wc + NEDGE;
    const int* cw_src = e_cw;          const int* cw_dst = e_cw + NEDGE;
    for (int t = 0; t < TT; ++t) {
        const float* xw = seqW + t * NH;
        const float* xc = fusedC + t * NH;
        gemm(stream, xw, HIDD, lw_kW, HIDD, 1, lw_kb, Pkw, HIDD, Nn, HIDD, HIDD, 0, ACT_NONE);
        gemm(stream, xw, HIDD, lw_vW, HIDD, 1, lw_vb, Pvw, HIDD, Nn, HIDD, HIDD, 0, ACT_NONE);
        gemm(stream, xw, HIDD, lw_qW, HIDD, 1, lw_qb, Pqw, HIDD, Nn, HIDD, HIDD, 0, ACT_NONE);
        gemm(stream, xc, HIDD, lc_kW, HIDD, 1, lc_kb, Pkc, HIDD, Nn, HIDD, HIDD, 0, ACT_NONE);
        gemm(stream, xc, HIDD, lc_vW, HIDD, 1, lc_vb, Pvc, HIDD, Nn, HIDD, HIDD, 0, ACT_NONE);
        gemm(stream, xc, HIDD, lc_qW, HIDD, 1, lc_qb, Pqc, HIDD, Nn, HIDD, HIDD, 0, ACT_NONE);
        // per-node relation transforms (hoisted off the edges): k@att, v@msg per head
        for (int h = 0; h < NHEAD; ++h) {
            gemm(stream, Pkw + h * DHEAD, HIDD, rwc_att + h * DHEAD * DHEAD, DHEAD, 0, nullptr, KTw + h * DHEAD, HIDD, Nn, DHEAD, DHEAD, 0, ACT_NONE);
            gemm(stream, Pvw + h * DHEAD, HIDD, rwc_msg + h * DHEAD * DHEAD, DHEAD, 0, nullptr, MTw + h * DHEAD, HIDD, Nn, DHEAD, DHEAD, 0, ACT_NONE);
            gemm(stream, Pkc + h * DHEAD, HIDD, rcw_att + h * DHEAD * DHEAD, DHEAD, 0, nullptr, KTc + h * DHEAD, HIDD, Nn, DHEAD, DHEAD, 0, ACT_NONE);
            gemm(stream, Pvc + h * DHEAD, HIDD, rcw_msg + h * DHEAD * DHEAD, DHEAD, 0, nullptr, MTc + h * DHEAD, HIDD, Nn, DHEAD, DHEAD, 0, ACT_NONE);
        }
        // relation wc: water -> city (q = city)
        attend(stream, wc_src, wc_dst, Pqc, KTw, MTw, rwc_mu, scoreB, smaxB, denB, aggB);
        gelu_kernel<<<(unsigned)((NH + 255) / 256), 256, 0, stream>>>(aggB, gelB, (int)NH);
        gemm(stream, gelB, HIDD, lc_aW, HIDD, 1, lc_ab, oT, HIDD, Nn, HIDD, HIDD, 0, ACT_NONE);
        hgt_skip_kernel<<<(unsigned)((NH + 255) / 256), 256, 0, stream>>>(oT, lc_sk, temb + t * HIDD, fusedC + t * NH, (int)NH);
        // relation cw: city -> water (q = water)
        attend(stream, cw_src, cw_dst, Pqw, KTc, MTc, rcw_mu, scoreB, smaxB, denB, aggB);
        gelu_kernel<<<(unsigned)((NH + 255) / 256), 256, 0, stream>>>(aggB, gelB, (int)NH);
        gemm(stream, gelB, HIDD, lw_aW, HIDD, 1, lw_ab, oT, HIDD, Nn, HIDD, HIDD, 0, ACT_NONE);
        hgt_skip_kernel<<<(unsigned)((NH + 255) / 256), 256, 0, stream>>>(oT, lw_sk, temb + t * HIDD, seqW + t * NH, (int)NH);
    }
    // ================= temporal MHA + LN (water then city) =================
    float* attnBuf = seqC;   // raw city GRU seq fully consumed -> reuse
    float* mhaTmp  = qkv;    // qkv consumed by attention kernel -> reuse head of region
    for (int which = 0; which < 2; ++which) {
        float* X = (which == 0) ? seqW : fusedC;
        gemm(stream, X, HIDD, ta_Wi, HIDD, 1, ta_bi, qkv, G3, TT * Nn, G3, HIDD, 0, ACT_NONE);
        tattn_kernel<<<dim3(Nn, NHEAD), 256, 0, stream>>>(qkv, attnBuf, Nn);
        gemm(stream, attnBuf, HIDD, ta_Wo, HIDD, 1, ta_bo, mhaTmp, HIDD, TT * Nn, HIDD, HIDD, 0, ACT_NONE);
        ln_kernel<<<TT * Nn, 32, 0, stream>>>(mhaTmp, X, n1_g, n1_b, X, TT * Nn);
    }
    // ================= final head =================
    const float* hwF = seqW   + (size_t)(TT - 1) * NH;
    const float* hcF = fusedC + (size_t)(TT - 1) * NH;
    gemm(stream, hwF, HIDD, xa_Wi,              HIDD, 1, xa_bi,       qx, HIDD, Nn, HIDD, HIDD, 0, ACT_NONE);
    gemm(stream, hcF, HIDD, xa_Wi + 128 * HIDD, HIDD, 1, xa_bi + 128, kx, HIDD, Nn, HIDD, HIDD, 0, ACT_NONE);
    gemm(stream, hcF, HIDD, xa_Wi + 256 * HIDD, HIDD, 1, xa_bi + 256, vx, HIDD, Nn, HIDD, HIDD, 0, ACT_NONE);
    cross_attn_kernel<<<dim3(Nn, NHEAD), 32, 0, stream>>>(qx, kx, vx, crs, Nn);
    gemm(stream, crs, HIDD, xa_Wo, HIDD, 1, xa_bo, hcx, HIDD, Nn, HIDD, HIDD, 0, ACT_NONE);
    // gate = sigmoid([hw_f | h_cross] @ Wg^T + bg)
    gemm(stream, hwF, HIDD, gt_W,        2 * HIDD, 1, gt_b,    gB, HIDD, Nn, HIDD, HIDD, 0, ACT_NONE);
    gemm(stream, hcx, HIDD, gt_W + HIDD, 2 * HIDD, 1, nullptr, gB, HIDD, Nn, HIDD, HIDD, 1, ACT_SIGM);
    gate_combine_kernel<<<(unsigned)((NH + 255) / 256), 256, 0, stream>>>(gB, hcx, hwF, hB, (int)NH);
    ln_kernel<<<Nn, 32, 0, stream>>>(hB, nullptr, n2_g, n2_b, hB, Nn);
    // FFN + residual + LN
    gemm(stream, hB, HIDD, ffn_W1, HIDD, 1, ffn_b1, f1, 2 * HIDD, Nn, 2 * HIDD, HIDD, 0, ACT_RELU);
    gemm(stream, f1, 2 * HIDD, ffn_W2, 2 * HIDD, 1, ffn_b2, f2, HIDD, Nn, HIDD, 2 * HIDD, 0, ACT_NONE);
    ln_kernel<<<Nn, 32, 0, stream>>>(f2, hB, n3_g, n3_b, hB, Nn);
    // prediction head -> d_out (N, 24)
    gemm(stream, hB, HIDD, pr_W1, HIDD, 1, pr_b1, p1, HIDD, Nn, HIDD, HIDD, 0, ACT_RELU);
    gemm(stream, p1, HIDD, pr_W2, HIDD, 1, pr_b2, (float*)d_out, OUTF, Nn, OUTF, HIDD, 0, ACT_NONE);
}
